// GraphNeuralNetwork_81140522156740
// MI455X (gfx1250) — matrix operand order repaired and checked by reading, not compiled, not run on hardware
//
#include <hip/hip_runtime.h>
#include <cstddef>

// ---------------------------------------------------------------------------
// GNN forward for MI455X (gfx1250, wave32).
// Edge MLP algebraically collapsed to rank-1 (edge_attr is [E,1]); gate GEMM
// hoisted to node level. Node GEMMs: v_wmma_f32_16x16x32_bf16 with
// pre-swizzled B fragments and unconditional b128 A loads.
// ---------------------------------------------------------------------------

#define IN_DIM 77
#define KPAD   96          // 77 padded to 3 K-chunks of 32
#define ND     64
#define ED     32
#define HD     128
#define NGRAPH 128
#define BN_EPS 1e-5f

typedef __attribute__((ext_vector_type(16))) __bf16 bf16x16;
typedef __attribute__((ext_vector_type(8)))  float  v8f;

static __device__ __forceinline__ void atomAddF(float* p, float v) {
  unsafeAtomicAdd(p, v);   // global_atomic_add_f32 on gfx1250
}

// ----------------------------- tiny utility --------------------------------
__global__ void zero_kernel(float* p, int n) {
  int i = blockIdx.x * blockDim.x + threadIdx.x;
  if (i < n) p[i] = 0.0f;
}

// pad x [N,77] -> xp [N,96] (zeros in 77..95) so the GEMM needs no K guards
__global__ __launch_bounds__(256)
void pad_x(const float* __restrict__ x, float* __restrict__ xp, int total) {
  int i = blockIdx.x * blockDim.x + threadIdx.x;
  if (i >= total) return;
  int r = i / KPAD;
  int c = i - r * KPAD;
  xp[i] = (c < IN_DIM) ? x[(size_t)r * IN_DIM + c] : 0.0f;
}

// ---------------------------------------------------------------------------
// Pack B [K,64] (row-major fp32) into WMMA B-fragment layout:
//   out[((kc*4 + nt)*32 + lane)*16 + e] = bf16(B[kc*32 + (e>>3)*16 + (lane>>4)*8 + (e&7)][nt*16 + (lane&15)])
// One contiguous 32-byte load per lane per fragment in the GEMM.
// ---------------------------------------------------------------------------
__global__ __launch_bounds__(256)
void pack_b(const float* __restrict__ B, int K, int kchunks,
            __bf16* __restrict__ out) {
  int i = blockIdx.x * blockDim.x + threadIdx.x;
  int total = kchunks * 4 * 32 * 16;
  if (i >= total) return;
  int e    = i & 15;
  int lane = (i >> 4) & 31;
  int nt   = (i >> 9) & 3;
  int kc   = i >> 11;
  int n    = (nt << 4) + (lane & 15);
  int k    = (kc << 5) + ((e >> 3) << 4) + ((lane >> 4) << 3) + (e & 7);
  out[i] = (k < K) ? (__bf16)B[(size_t)k * ND + n] : (__bf16)0.0f;
}

// ---------------------------------------------------------------------------
// Precompute per-layer rank-1 edge vectors:
//   u[l] = eemb_row @ We[l] @ Wg_bot[l]
//   v[l] = (eemb_b @ We[l] + be[l]) @ Wg_bot[l] + bg[l]
// ---------------------------------------------------------------------------
__global__ void precompute_uv(const float* __restrict__ eembW,
                              const float* __restrict__ eembB,
                              const float* __restrict__ We,
                              const float* __restrict__ be,
                              const float* __restrict__ Wg,
                              const float* __restrict__ bg,
                              float* __restrict__ uv) {
  const int l = blockIdx.x;
  const int j = threadIdx.x;                 // 0..63
  __shared__ float t[ND], tb[ND];
  float a = 0.f, b = 0.f;
  for (int k = 0; k < ED; ++k) {
    float w = We[(size_t)l * ED * ND + (size_t)k * ND + j];
    a += eembW[k] * w;
    b += eembB[k] * w;
  }
  t[j]  = a;
  tb[j] = b + be[l * ND + j];
  __syncthreads();
  float u = 0.f, v = 0.f;
  for (int jj = 0; jj < ND; ++jj) {
    float w = Wg[(size_t)l * 2 * ND * ND + (size_t)(ND + jj) * ND + j];
    u += t[jj]  * w;
    v += tb[jj] * w;
  }
  uv[l * 2 * ND + j]      = u;
  uv[l * 2 * ND + ND + j] = v + bg[l * ND + j];
}

// ------------------- A-fragment loader (no predication) --------------------
// k indices of the 16 elements are two contiguous 8-float runs starting at
// hi*8 and 16+hi*8 within the 32-wide chunk -> 4 unconditional float4 loads.
static __device__ __forceinline__ bf16x16 load_afrag(const float* __restrict__ ak) {
  float4 a0 = *reinterpret_cast<const float4*>(ak);
  float4 a1 = *reinterpret_cast<const float4*>(ak + 4);
  float4 a2 = *reinterpret_cast<const float4*>(ak + 16);
  float4 a3 = *reinterpret_cast<const float4*>(ak + 20);
  bf16x16 af;
  af[0]  = (__bf16)a0.x; af[1]  = (__bf16)a0.y; af[2]  = (__bf16)a0.z; af[3]  = (__bf16)a0.w;
  af[4]  = (__bf16)a1.x; af[5]  = (__bf16)a1.y; af[6]  = (__bf16)a1.z; af[7]  = (__bf16)a1.w;
  af[8]  = (__bf16)a2.x; af[9]  = (__bf16)a2.y; af[10] = (__bf16)a2.z; af[11] = (__bf16)a2.w;
  af[12] = (__bf16)a3.x; af[13] = (__bf16)a3.y; af[14] = (__bf16)a3.z; af[15] = (__bf16)a3.w;
  return af;
}

// ---------------------------------------------------------------------------
// C[M,64] = A[M,KC*32](lda) @ B + bias.  B pre-swizzled (pack_b).
// One wave32 per 16x64 tile; M must be a multiple of 16 (N = 3125*16).
// ---------------------------------------------------------------------------
template <int KC>
__global__ __launch_bounds__(256)
void gemm_wmma(const float* __restrict__ A, int lda,
               const __bf16* __restrict__ Bp, const float* __restrict__ bias,
               float* __restrict__ C, int Mt) {
  const int lane = threadIdx.x & 31;
  const int wave = threadIdx.x >> 5;
  const int mt   = blockIdx.x * 8 + wave;
  if (mt >= Mt) return;                       // wave-uniform: EXEC stays all-ones
  const int ln15 = lane & 15;
  const int hi   = lane >> 4;
  const float* arow = A + (size_t)(mt * 16 + ln15) * lda + hi * 8;
  const bf16x16* bp = reinterpret_cast<const bf16x16*>(Bp);

  v8f acc[4] = {v8f{0}, v8f{0}, v8f{0}, v8f{0}};
#pragma unroll
  for (int kc = 0; kc < KC; ++kc) {
    bf16x16 af = load_afrag(arow + kc * 32);
#pragma unroll
    for (int nt = 0; nt < 4; ++nt) {
      bf16x16 bf = bp[((kc << 2) + nt) * 32 + lane];
      acc[nt] = __builtin_amdgcn_wmma_f32_16x16x32_bf16(
          false, af, false, bf, (short)0, acc[nt], false, false);
    }
  }
  const int mr = mt * 16;
#pragma unroll
  for (int nt = 0; nt < 4; ++nt) {
    const int n  = (nt << 4) + ln15;
    const float bv = bias ? bias[n] : 0.0f;
#pragma unroll
    for (int r = 0; r < 8; ++r)
      C[(size_t)(mr + r + hi * 8) * ND + n] = acc[nt][r] + bv;
  }
}

// ---------------------------------------------------------------------------
// Dual-output node GEMM: C1 = A@B1 + bias1, C2 = A@B2.
// Reuses each A fragment for 8 WMMAs (halves A traffic vs two launches).
// ---------------------------------------------------------------------------
template <int KC>
__global__ __launch_bounds__(256)
void gemm_wmma_dual(const float* __restrict__ A, int lda,
                    const __bf16* __restrict__ Bp1, const float* __restrict__ bias1,
                    float* __restrict__ C1,
                    const __bf16* __restrict__ Bp2, float* __restrict__ C2,
                    int Mt) {
  const int lane = threadIdx.x & 31;
  const int wave = threadIdx.x >> 5;
  const int mt   = blockIdx.x * 8 + wave;
  if (mt >= Mt) return;
  const int ln15 = lane & 15;
  const int hi   = lane >> 4;
  const float* arow = A + (size_t)(mt * 16 + ln15) * lda + hi * 8;
  const bf16x16* bp1 = reinterpret_cast<const bf16x16*>(Bp1);
  const bf16x16* bp2 = reinterpret_cast<const bf16x16*>(Bp2);

  v8f acc1[4] = {v8f{0}, v8f{0}, v8f{0}, v8f{0}};
  v8f acc2[4] = {v8f{0}, v8f{0}, v8f{0}, v8f{0}};
#pragma unroll
  for (int kc = 0; kc < KC; ++kc) {
    bf16x16 af = load_afrag(arow + kc * 32);
#pragma unroll
    for (int nt = 0; nt < 4; ++nt) {
      bf16x16 b1 = bp1[((kc << 2) + nt) * 32 + lane];
      acc1[nt] = __builtin_amdgcn_wmma_f32_16x16x32_bf16(
          false, af, false, b1, (short)0, acc1[nt], false, false);
      bf16x16 b2 = bp2[((kc << 2) + nt) * 32 + lane];
      acc2[nt] = __builtin_amdgcn_wmma_f32_16x16x32_bf16(
          false, af, false, b2, (short)0, acc2[nt], false, false);
    }
  }
  const int mr = mt * 16;
#pragma unroll
  for (int nt = 0; nt < 4; ++nt) {
    const int n  = (nt << 4) + ln15;
    const float bv = bias1[n];
#pragma unroll
    for (int r = 0; r < 8; ++r) {
      size_t row = (size_t)(mr + r + hi * 8);
      C1[row * ND + n] = acc1[nt][r] + bv;
      C2[row * ND + n] = acc2[nt][r];
    }
  }
}

// ---------------------------------------------------------------------------
// Per-edge fused gate + message + scatter-add:
//   gate = sigmoid(p[src] + edge_attr*u + v);  acc[dst] += gate * h[src]
// 16 threads per edge, 4 columns each (float4 gathers, f32 atomics to L2).
// ---------------------------------------------------------------------------
__global__ __launch_bounds__(256)
void edge_kernel(const int* __restrict__ src, const int* __restrict__ dst,
                 const float* __restrict__ eattr,
                 const float* __restrict__ p, const float* __restrict__ h,
                 const float* __restrict__ uv_l,
                 float* __restrict__ acc, int E) {
  __shared__ float su[ND], sv[ND];
  if (threadIdx.x < ND)            su[threadIdx.x]      = uv_l[threadIdx.x];
  else if (threadIdx.x < 2 * ND)   sv[threadIdx.x - ND] = uv_l[threadIdx.x];
  __syncthreads();

  const int gid = blockIdx.x * blockDim.x + threadIdx.x;
  if (gid >= E * 16) return;
  const int e  = gid >> 4;
  const int c0 = (gid & 15) << 2;
  const int s  = src[e];
  const int d  = dst[e];
  const float a = eattr[e];

  const float4 pv = *reinterpret_cast<const float4*>(p + (size_t)s * ND + c0);
  const float4 hv = *reinterpret_cast<const float4*>(h + (size_t)s * ND + c0);

  float g;
  float* out = acc + (size_t)d * ND + c0;
  g = 1.0f / (1.0f + __expf(-(pv.x + a * su[c0 + 0] + sv[c0 + 0])));
  atomAddF(out + 0, g * hv.x);
  g = 1.0f / (1.0f + __expf(-(pv.y + a * su[c0 + 1] + sv[c0 + 1])));
  atomAddF(out + 1, g * hv.y);
  g = 1.0f / (1.0f + __expf(-(pv.z + a * su[c0 + 2] + sv[c0 + 2])));
  atomAddF(out + 2, g * hv.z);
  g = 1.0f / (1.0f + __expf(-(pv.w + a * su[c0 + 3] + sv[c0 + 3])));
  atomAddF(out + 3, g * hv.w);
}

// ---------------------------------------------------------------------------
// BatchNorm column stats: stats[0:64]=sum, stats[64:128]=sumsq over N rows.
// ---------------------------------------------------------------------------
__global__ __launch_bounds__(256)
void stats_kernel(const float* __restrict__ acc, float* __restrict__ stats, int N) {
  __shared__ float s_sum[4][ND], s_sq[4][ND];
  const int c   = threadIdx.x & 63;
  const int sub = threadIdx.x >> 6;          // 0..3
  float sum = 0.f, sq = 0.f;
  for (int r = blockIdx.x * 4 + sub; r < N; r += gridDim.x * 4) {
    float x = acc[(size_t)r * ND + c];
    sum += x; sq += x * x;
  }
  s_sum[sub][c] = sum;
  s_sq[sub][c]  = sq;
  __syncthreads();
  if (threadIdx.x < ND) {
    float ts = s_sum[0][c] + s_sum[1][c] + s_sum[2][c] + s_sum[3][c];
    float tq = s_sq[0][c]  + s_sq[1][c]  + s_sq[2][c]  + s_sq[3][c];
    atomAddF(&stats[c], ts);
    atomAddF(&stats[ND + c], tq);
  }
}

// h = relu(bn(acc))
__global__ __launch_bounds__(256)
void bn_apply(const float* __restrict__ acc, const float* __restrict__ stats,
              const float* __restrict__ gamma, const float* __restrict__ beta,
              float* __restrict__ h, int total, float invN) {
  int i = blockIdx.x * blockDim.x + threadIdx.x;
  if (i >= total) return;
  int c = i & 63;
  float mean = stats[c] * invN;
  float var  = stats[ND + c] * invN - mean * mean;
  float sc   = rsqrtf(var + BN_EPS) * gamma[c];
  float sh   = beta[c] - mean * sc;
  float y = acc[i] * sc + sh;
  h[i] = y > 0.f ? y : 0.f;
}

// segment-sum pooling over graph ids
__global__ __launch_bounds__(256)
void pool_kernel(const float* __restrict__ h, const int* __restrict__ batch,
                 float* __restrict__ pooled, float* __restrict__ counts, int total) {
  int i = blockIdx.x * blockDim.x + threadIdx.x;
  if (i >= total) return;
  int c = i & 63;
  int r = i >> 6;
  int g = batch[r];
  atomAddF(&pooled[(size_t)g * ND + c], h[i]);
  if (c == 0) atomAddF(&counts[g], 1.0f);
}

// ---------------------------------------------------------------------------
// Final: mean-pool divide, [128,64]@[64,128]+fb, BN over 128 rows, ReLU.
// ---------------------------------------------------------------------------
__global__ __launch_bounds__(256)
void final_kernel(const float* __restrict__ pooled, const float* __restrict__ counts,
                  const float* __restrict__ fW, const float* __restrict__ fb,
                  const float* __restrict__ fgamma, const float* __restrict__ fbeta,
                  float* __restrict__ out) {
  __shared__ float pl[NGRAPH * ND];          // 32 KB of the 320 KB LDS
  __shared__ float red[4 * HD];
  const int t = threadIdx.x;
  for (int i = t; i < NGRAPH * ND; i += 256) {
    float cnt = counts[i >> 6];
    cnt = cnt < 1.f ? 1.f : cnt;
    pl[i] = pooled[i] / cnt;
  }
  __syncthreads();

  const int c    = t & (HD - 1);             // column 0..127
  const int half = t >> 7;                   // row half 0/1
  float z[64];
  float sum = 0.f, sq = 0.f;
  for (int rr = 0; rr < 64; ++rr) {
    int r = half * 64 + rr;
    float a = fb[c];
    for (int k = 0; k < ND; ++k) a += pl[r * ND + k] * fW[(size_t)k * HD + c];
    z[rr] = a; sum += a; sq += a * a;
  }
  red[half * HD + c]          = sum;
  red[2 * HD + half * HD + c] = sq;
  __syncthreads();
  float ts = red[c] + red[HD + c];
  float tq = red[2 * HD + c] + red[3 * HD + c];
  float mean = ts * (1.0f / NGRAPH);
  float var  = tq * (1.0f / NGRAPH) - mean * mean;
  float sc   = rsqrtf(var + BN_EPS) * fgamma[c];
  float sh   = fbeta[c] - mean * sc;
  for (int rr = 0; rr < 64; ++rr) {
    int r = half * 64 + rr;
    float y = z[rr] * sc + sh;
    out[(size_t)r * HD + c] = y > 0.f ? y : 0.f;
  }
}

// ---------------------------------------------------------------------------
extern "C" void kernel_launch(void* const* d_in, const int* in_sizes, int n_in,
                              void* d_out, int out_size, void* d_ws, size_t ws_size,
                              hipStream_t stream) {
  const float* x      = (const float*)d_in[0];
  const int*   eidx   = (const int*)  d_in[1];
  const float* eattr  = (const float*)d_in[2];
  const int*   batch  = (const int*)  d_in[3];
  const float* embW   = (const float*)d_in[4];
  const float* embB   = (const float*)d_in[5];
  const float* eembW  = (const float*)d_in[6];
  const float* eembB  = (const float*)d_in[7];
  const float* Wn     = (const float*)d_in[8];
  const float* bnb    = (const float*)d_in[9];
  const float* We     = (const float*)d_in[10];
  const float* be     = (const float*)d_in[11];
  const float* Wg     = (const float*)d_in[12];
  const float* bg     = (const float*)d_in[13];
  const float* gamma  = (const float*)d_in[14];
  const float* beta   = (const float*)d_in[15];
  const float* fW     = (const float*)d_in[16];
  const float* fb     = (const float*)d_in[17];
  const float* fgamma = (const float*)d_in[18];
  const float* fbeta  = (const float*)d_in[19];

  const int N = in_sizes[0] / IN_DIM;
  const int E = in_sizes[2];
  const int L = 3;

  // ws layout (floats): h | p | acc | xp | uv | stats | pooled | counts | Bpk...
  float* ws     = (float*)d_ws;
  float* h      = ws;
  float* p      = h   + (size_t)N * ND;
  float* acc    = p   + (size_t)N * ND;
  float* xp     = acc + (size_t)N * ND;
  float* uv     = xp  + (size_t)N * KPAD;
  float* stats  = uv + L * 2 * ND;
  float* pooled = stats + 2 * ND;
  float* counts = pooled + NGRAPH * ND;
  __bf16* bpk0  = (__bf16*)(counts + NGRAPH);   // embed B: 3*4*32*16 bf16
  __bf16* bpk1  = bpk0 + 3 * 2048;              // Wn[l]:   2*4*32*16
  __bf16* bpk2  = bpk1 + 2 * 2048;              // Wg_top[l]

  const int Mt        = (N + 15) / 16;          // 3125 (N multiple of 16)
  const int gemm_grid = (Mt + 7) / 8;
  const int elems     = N * ND;
  const int egrid     = (elems + 255) / 256;

  precompute_uv<<<L, ND, 0, stream>>>(eembW, eembB, We, be, Wg, bg, uv);
  zero_kernel<<<(NGRAPH * ND + NGRAPH + 255) / 256, 256, 0, stream>>>(pooled, NGRAPH * ND + NGRAPH);
  pad_x<<<(N * KPAD + 255) / 256, 256, 0, stream>>>(x, xp, N * KPAD);
  pack_b<<<(3 * 2048 + 255) / 256, 256, 0, stream>>>(embW, IN_DIM, 3, bpk0);

  // h = x @ emb_W + emb_b
  gemm_wmma<3><<<gemm_grid, 256, 0, stream>>>(xp, KPAD, bpk0, embB, h, Mt);

  for (int l = 0; l < L; ++l) {
    pack_b<<<(2 * 2048 + 255) / 256, 256, 0, stream>>>(Wn + (size_t)l * ND * ND, ND, 2, bpk1);
    pack_b<<<(2 * 2048 + 255) / 256, 256, 0, stream>>>(Wg + (size_t)l * 2 * ND * ND, ND, 2, bpk2);
    // acc = h@Wn[l] + bn_b[l]  (edge kernel adds on top);  p = h@Wg_top[l]
    gemm_wmma_dual<2><<<gemm_grid, 256, 0, stream>>>(h, ND, bpk1, bnb + l * ND, acc,
                                                     bpk2, p, Mt);
    zero_kernel<<<1, 2 * ND, 0, stream>>>(stats, 2 * ND);
    edge_kernel<<<(E * 16 + 255) / 256, 256, 0, stream>>>(eidx, eidx + E, eattr, p, h,
                                                          uv + l * 2 * ND, acc, E);
    stats_kernel<<<256, 256, 0, stream>>>(acc, stats, N);
    bn_apply<<<egrid, 256, 0, stream>>>(acc, stats, gamma + l * ND, beta + l * ND,
                                        h, elems, 1.0f / (float)N);
  }

  pool_kernel<<<egrid, 256, 0, stream>>>(h, batch, pooled, counts, elems);
  final_kernel<<<1, 256, 0, stream>>>(pooled, counts, fW, fb, fgamma, fbeta, (float*)d_out);
}
